// Seq2Seq_12309376270950
// MI455X (gfx1250) — compile-verified, matrix operand order checked by reading, not run on hardware
//
#include <hip/hip_runtime.h>

// ---------------------------------------------------------------------------
// Seq2Seq (GRU encoder/decoder + vocab projection + log_softmax) for gfx1250.
// V=32000, H=512, B=32, T=64.  GEMMs via v_wmma_f32_16x16x32_bf16; tiles are
// staged into LDS with TENSOR_LOAD_TO_LDS (TDM) + GLOBAL_LOAD_ASYNC_TO_LDS.
// ---------------------------------------------------------------------------

#define Vv 32000
#define Hh 512
#define Bb 32
#define Tt 64

typedef __attribute__((ext_vector_type(16))) __bf16 v16bf;
typedef __attribute__((ext_vector_type(8)))  __bf16 v8bf;
typedef __attribute__((ext_vector_type(8)))  float  v8f;
typedef int av4i __attribute__((vector_size(16)));   // matches builtin's V4i

#if defined(__HIP_DEVICE_COMPILE__) && __has_builtin(__builtin_amdgcn_global_load_async_to_lds_b128)
#define USE_ASYNC_LDS 1
#else
#define USE_ASYNC_LDS 0
#endif
#if defined(__HIP_DEVICE_COMPILE__) && __has_builtin(__builtin_amdgcn_tensor_load_to_lds)
#define USE_TDM 1
#else
#define USE_TDM 0
#endif

// ---- WMMA helpers ----------------------------------------------------------

__device__ __forceinline__ v8f wmma_bf16(v16bf a, v16bf b, v8f c) {
  // D = A(16x32) * B(32x16) + C(16x16), fp32 accumulate
  return __builtin_amdgcn_wmma_f32_16x16x32_bf16(
      /*neg_a=*/false, a, /*neg_b=*/false, b,
      /*c_mod=*/(short)0, c, /*reuse_a=*/false, /*reuse_b=*/false);
}

// Load one 16x32 bf16 A/B fragment from a row-major buffer (LDS or global).
// ISA 16-bit A layout: lanes 0-15 hold K {0..7,16..23}, lanes 16-31 hold
// K {8..15,24..31}; row = row0 + (lane&15).
__device__ __forceinline__ v16bf frag_ld(const __bf16* base, int row0,
                                         int stride, int lane) {
  const __bf16* p = base + (size_t)(row0 + (lane & 15)) * stride
                        + ((lane >> 4) << 3);
  v8bf lo = *(const v8bf*)p;         // K + {0..7}  (or {8..15})
  v8bf hi = *(const v8bf*)(p + 16);  // K + {16..23} (or {24..31})
  return __builtin_shufflevector(lo, hi, 0, 1, 2, 3, 4, 5, 6, 7,
                                 8, 9, 10, 11, 12, 13, 14, 15);
}

// ---- async global->LDS (per-lane DMA, ASYNCcnt) ----------------------------
#if USE_ASYNC_LDS
template <int OFS>
__device__ __forceinline__ void async_g2l16(const void* g, void* l) {
  __builtin_amdgcn_global_load_async_to_lds_b128(
      (__attribute__((address_space(1))) av4i*)(av4i*)(void*)g,
      (__attribute__((address_space(3))) av4i*)(av4i*)l, OFS, 0);
}
#endif

__device__ __forceinline__ void wait_async0() {
#if USE_ASYNC_LDS
#if __has_builtin(__builtin_amdgcn_s_wait_asynccnt)
  __builtin_amdgcn_s_wait_asynccnt(0);
#else
  asm volatile("s_wait_asynccnt 0x0" ::: "memory");
#endif
#endif
}

// ---- TDM: DMA one 2D bf16 tile (tile_d1 rows x tile_d0 elems) to LDS -------
// LDS rows are padded: 64B of data then 16B pad (pad_interval=16 DWORDs,
// pad_amount=4 DWORDs) to match the 40-half LDS row stride used by frag_ld.
#if USE_TDM
__device__ __forceinline__ unsigned lds_off(const void* p) {
  return (unsigned)(unsigned long long)
      (__attribute__((address_space(3))) const void*)p;
}

__device__ __forceinline__ void tdm_load_tile_bf16(
    unsigned lds_addr, const void* gaddr, unsigned tensor_d0,
    unsigned tensor_d1, unsigned tile_d0, unsigned tile_d1,
    unsigned stride_d0) {
  typedef unsigned int u32x4 __attribute__((ext_vector_type(4)));
  typedef int i32x4 __attribute__((ext_vector_type(4)));
  typedef int i32x8 __attribute__((ext_vector_type(8)));
  unsigned long long ga = (unsigned long long)gaddr;
  u32x4 g0;
  g0.x = 1u;                                   // count=1, user descriptor
  g0.y = lds_addr;                             // LDS dest (bytes)
  g0.z = (unsigned)(ga & 0xffffffffu);         // global_addr[31:0]
  g0.w = (unsigned)((ga >> 32) & 0x1ffffffu)   // global_addr[56:32]
         | (2u << 30);                         // type=2 ("image")
  i32x8 g1;
  g1[0] = (1 << 16)                            // data_size = 2 bytes
        | (1 << 20)                            // pad_enable
        | (3 << 22)                            // pad_interval: 16 DWORDs
        | (3 << 25);                           // pad_amount:   4 DWORDs
  g1[1] = (int)((tensor_d0 & 0xffffu) << 16);  // tensor_dim0[15:0]
  g1[2] = (int)((tensor_d0 >> 16) | ((tensor_d1 & 0xffffu) << 16));
  g1[3] = (int)((tensor_d1 >> 16) | (tile_d0 << 16));   // tile_dim0
  g1[4] = (int)(tile_d1 & 0xffffu);            // tile_dim1 (tile_dim2=0)
  g1[5] = (int)stride_d0;                      // tensor_dim0_stride[31:0]
  g1[6] = 0;
  g1[7] = 0;
  i32x4 g2 = (i32x4)0, g3 = (i32x4)0;
#if __clang_major__ >= 23
  i32x8 g4 = (i32x8)0;
  __builtin_amdgcn_tensor_load_to_lds(g0, g1, g2, g3, g4, 0);
#else
  __builtin_amdgcn_tensor_load_to_lds(g0, g1, g2, g3, 0);
#endif
}
#endif  // USE_TDM

// ---- device-scope barrier across the scan workgroups -----------------------
__device__ __forceinline__ void gbar(unsigned* cnt, unsigned* gen,
                                     unsigned nwg) {
  __syncthreads();
  if (threadIdx.x == 0) {
    __threadfence();
    volatile unsigned* vgen = (volatile unsigned*)gen;
    unsigned g = *vgen;
    if (atomicAdd(cnt, 1u) == nwg - 1u) {
      atomicExch(cnt, 0u);
      __threadfence();
      atomicAdd(gen, 1u);
    } else {
      while (*vgen == g) __builtin_amdgcn_s_sleep(1);
    }
    __threadfence();
  }
  __syncthreads();
}

// ---------------------------------------------------------------------------
// fp32 -> bf16 conversion (weights)
// ---------------------------------------------------------------------------
__global__ void cvt_k(const float* __restrict__ s, __bf16* __restrict__ d,
                      int n) {
  int gid = blockIdx.x * blockDim.x + threadIdx.x;
  int gsz = gridDim.x * blockDim.x;
  for (int i = gid; i < n; i += gsz) d[i] = (__bf16)s[i];
}

// embedding gather -> bf16 A matrix, rows m = t*B + b  (t-major for scan)
__global__ void embed_enc_k(const int* __restrict__ seqs,
                            const float* __restrict__ emb,
                            __bf16* __restrict__ out, int n) {
  int gid = blockIdx.x * blockDim.x + threadIdx.x;
  if (gid >= n) return;
  int k = gid & (Hh - 1);
  int m = gid >> 9;           // H = 512
  int b = m & (Bb - 1);
  int t = m >> 5;             // B = 32
  out[gid] = (__bf16)emb[(size_t)seqs[b * Tt + t] * Hh + k];
}

// decoder input: relu(concat(dec_emb[target], thought)) -> bf16, rows t*B+b
__global__ void concat_relu_k(const int* __restrict__ seqs,
                              const float* __restrict__ emb,
                              const float* __restrict__ thought,
                              __bf16* __restrict__ out, int n) {
  int gid = blockIdx.x * blockDim.x + threadIdx.x;
  if (gid >= n) return;
  int k = gid & (2 * Hh - 1);
  int m = gid >> 10;          // 2H = 1024
  int b = m & (Bb - 1);
  int t = m >> 5;
  float v = (k < Hh) ? emb[(size_t)seqs[b * Tt + t] * Hh + k]
                     : thought[(size_t)b * Hh + (k - Hh)];
  out[gid] = (__bf16)fmaxf(v, 0.0f);
}

// ---------------------------------------------------------------------------
// GEMM: C[M,N] = A[M,K](bf16) @ W[N,K](bf16)^T + bias[N], C fp32.
// 256 threads (8 waves), 128x128 tile, K step 32.
// A tile staged via GLOBAL_LOAD_ASYNC_TO_LDS_B128 (per-lane DMA);
// W tile staged via one TENSOR_LOAD_TO_LDS (TDM) per K slice.
// M, N multiples of 128; K multiple of 32.
// ---------------------------------------------------------------------------
__global__ __launch_bounds__(256) void gemm_bf16_k(
    const __bf16* __restrict__ A, const __bf16* __restrict__ W,
    const float* __restrict__ bias, float* __restrict__ C,
    int M, int N, int K) {
  __shared__ __bf16 sA[128 * 40];   // 32-half rows + 8-half pad
  __shared__ __bf16 sW[128 * 40];
  const int tid = threadIdx.x;
  const int lane = tid & 31;
  const int wave = tid >> 5;
  const int bm = blockIdx.y * 128;
  const int bn = blockIdx.x * 128;
  const int wm = (wave & 3) * 32;   // 4 waves along M
  const int wn = (wave >> 2) * 64;  // 2 waves along N
  const int lrow = tid >> 1;        // 0..127
  const int lcol = (tid & 1) * 16;  // 0 or 16
  const __bf16* ga = A + (size_t)(bm + lrow) * K + lcol;
  const __bf16* gw = W + (size_t)(bn + lrow) * K + lcol;

  v8f acc[2][4] = {};
  for (int k0 = 0; k0 < K; k0 += 32) {
    // ---- stage A tile (per-lane async DMA straight into LDS) ----
#if USE_ASYNC_LDS
    async_g2l16<0>(ga + k0, &sA[lrow * 40 + lcol]);
    async_g2l16<16>(ga + k0, &sA[lrow * 40 + lcol]);  // inst offset -> both
#else
    {
      v8bf a0 = *(const v8bf*)(ga + k0);
      v8bf a1 = *(const v8bf*)(ga + k0 + 8);
      *(v8bf*)&sA[lrow * 40 + lcol] = a0;
      *(v8bf*)&sA[lrow * 40 + lcol + 8] = a1;
    }
#endif
    // ---- stage W tile (one TDM descriptor moves the whole 128x32 tile) ----
#if USE_TDM
    if (tid == 0) {
      tdm_load_tile_bf16(lds_off(&sW[0]),
                         W + (size_t)bn * K + k0,
                         /*tensor_d0=*/(unsigned)K, /*tensor_d1=*/(unsigned)N,
                         /*tile_d0=*/32u, /*tile_d1=*/128u,
                         /*stride_d0=*/(unsigned)K);
      __builtin_amdgcn_s_wait_tensorcnt(0);
    }
#elif USE_ASYNC_LDS
    async_g2l16<0>(gw + k0, &sW[lrow * 40 + lcol]);
    async_g2l16<16>(gw + k0, &sW[lrow * 40 + lcol]);
#else
    {
      v8bf w0 = *(const v8bf*)(gw + k0);
      v8bf w1 = *(const v8bf*)(gw + k0 + 8);
      *(v8bf*)&sW[lrow * 40 + lcol] = w0;
      *(v8bf*)&sW[lrow * 40 + lcol + 8] = w1;
    }
#endif
    wait_async0();
    __syncthreads();

    v16bf af[2], wf[4];
#pragma unroll
    for (int i = 0; i < 2; ++i) af[i] = frag_ld(sA, wm + i * 16, 40, lane);
#pragma unroll
    for (int j = 0; j < 4; ++j) wf[j] = frag_ld(sW, wn + j * 16, 40, lane);
#pragma unroll
    for (int i = 0; i < 2; ++i)
#pragma unroll
      for (int j = 0; j < 4; ++j) acc[i][j] = wmma_bf16(af[i], wf[j], acc[i][j]);
    __syncthreads();
  }
  // C/D layout: VGPR r -> row r (lanes 0-15) or 8+r (lanes 16-31); col=lane&15
#pragma unroll
  for (int i = 0; i < 2; ++i) {
    const int rbase = bm + wm + i * 16 + ((lane >> 4) << 3);
#pragma unroll
    for (int j = 0; j < 4; ++j) {
      const int col = bn + wn + j * 16 + (lane & 15);
      const float bv = bias ? bias[col] : 0.0f;
#pragma unroll
      for (int r = 0; r < 8; ++r)
        C[(size_t)(rbase + r) * N + col] = acc[i][j][r] + bv;
    }
  }
}

// ---------------------------------------------------------------------------
// Cooperative GRU scan: 32 WGs x 192 threads (6 waves). WG w owns hidden
// columns j in [16w, 16w+16); per step computes hp = h @ Whh^T + bhh for its
// gate triplet {j, H+j, 2H+j} via WMMA (Whh slice resident in LDS as bf16),
// runs the gate math in fp32, publishes bf16 h to a ping-pong buffer, then
// crosses a device-scope barrier.  xp: [T][B][3H] fp32.
// ---------------------------------------------------------------------------
__global__ __launch_bounds__(192) void gru_scan_k(
    const float* __restrict__ xp, const float* __restrict__ Whh,
    const float* __restrict__ bhh, __bf16* __restrict__ hbuf,
    float* __restrict__ thought, __bf16* __restrict__ seq_out,
    unsigned* __restrict__ bar_cnt, unsigned* __restrict__ bar_gen,
    int Tsteps) {
  __shared__ __bf16 sW[48 * 520];   // 48 Whh rows x 512 (+8 pad) bf16 ~ 49.9KB
  __shared__ float hp[32 * 48];     // per-step recurrent pre-activations
  __shared__ float h32[32 * 16];    // fp32 h for this WG's 16 columns
  const int tid = threadIdx.x;
  const int wg = blockIdx.x;        // 0..31
  const unsigned nwg = gridDim.x;
  const int lane = tid & 31;
  const int wave = tid >> 5;        // 0..5
  const int mi = wave & 1;          // M tile (batch 0-15 / 16-31)
  const int ni = wave >> 1;         // gate group 0..2 (r/z/n)
  const int j0 = wg * 16;

  // Stage Whh rows {g*512 + j0 + rr} as bf16 (once for all 64 steps).
  for (int idx = tid; idx < 48 * 512; idx += 192) {
    int lr = idx >> 9, k = idx & 511;
    int g = lr >> 4, rr = lr & 15;
    sW[lr * 520 + k] = (__bf16)Whh[(size_t)(g * Hh + j0 + rr) * Hh + k];
  }
  // h0 = 0 (bf16 publish slice + fp32 private slice)
  for (int idx = tid; idx < 512; idx += 192) {
    int b = idx >> 4, jj = idx & 15;
    hbuf[(size_t)b * Hh + j0 + jj] = (__bf16)0.0f;
    h32[b * 16 + jj] = 0.0f;
  }
  __threadfence();
  gbar(bar_cnt, bar_gen, nwg);

  for (int t = 0; t < Tsteps; ++t) {
    const __bf16* hc = hbuf + (size_t)(t & 1) * (Bb * Hh);
    v8f acc = {};
#pragma unroll
    for (int k0 = 0; k0 < Hh; k0 += 32) {
      v16bf af = frag_ld(hc + k0, mi * 16, Hh, lane);   // h from L2 (bf16)
      v16bf wf = frag_ld(sW + k0, ni * 16, 520, lane);  // Whh from LDS
      acc = wmma_bf16(af, wf, acc);
    }
    {
      const int col = ni * 16 + (lane & 15);
      const int rbase = mi * 16 + ((lane >> 4) << 3);
#pragma unroll
      for (int r = 0; r < 8; ++r) hp[(rbase + r) * 48 + col] = acc[r];
    }
    __syncthreads();

    __bf16* hn_out = hbuf + (size_t)((t + 1) & 1) * (Bb * Hh);
    for (int idx = tid; idx < 512; idx += 192) {
      int b = idx >> 4, jj = idx & 15;
      int j = j0 + jj;
      const float* xpt = xp + ((size_t)t * Bb + b) * (3 * Hh);
      float hr = hp[b * 48 + jj]      + bhh[j];
      float hz = hp[b * 48 + 16 + jj] + bhh[Hh + j];
      float hn = hp[b * 48 + 32 + jj] + bhh[2 * Hh + j];
      float r = 1.0f / (1.0f + expf(-(xpt[j] + hr)));
      float z = 1.0f / (1.0f + expf(-(xpt[Hh + j] + hz)));
      float n = tanhf(xpt[2 * Hh + j] + r * hn);
      float hnew = (1.0f - z) * n + z * h32[b * 16 + jj];
      h32[b * 16 + jj] = hnew;
      hn_out[(size_t)b * Hh + j] = (__bf16)hnew;
      if (seq_out) seq_out[((size_t)b * Tsteps + t) * Hh + j] = (__bf16)hnew;
      if (thought && t == Tsteps - 1) thought[(size_t)b * Hh + j] = hnew;
    }
    __threadfence();
    gbar(bar_cnt, bar_gen, nwg);
  }
}

// ---------------------------------------------------------------------------
// log_softmax over rows of 32000, in place. One block per (b,t) row.
// ---------------------------------------------------------------------------
__global__ __launch_bounds__(256) void logsoftmax_k(float* __restrict__ C,
                                                    int N) {
  __shared__ float red[256];
  float* row = C + (size_t)blockIdx.x * N;
  const int tid = threadIdx.x;
  float m = -3.4e38f;
  for (int i = tid; i < N; i += 256) m = fmaxf(m, row[i]);
  red[tid] = m;
  __syncthreads();
  for (int s = 128; s > 0; s >>= 1) {
    if (tid < s) red[tid] = fmaxf(red[tid], red[tid + s]);
    __syncthreads();
  }
  m = red[0];
  __syncthreads();
  float sum = 0.0f;
  for (int i = tid; i < N; i += 256) sum += expf(row[i] - m);
  red[tid] = sum;
  __syncthreads();
  for (int s = 128; s > 0; s >>= 1) {
    if (tid < s) red[tid] += red[tid + s];
    __syncthreads();
  }
  const float lg = m + logf(red[0]);
  for (int i = tid; i < N; i += 256) row[i] -= lg;
}

// ---------------------------------------------------------------------------
extern "C" void kernel_launch(void* const* d_in, const int* in_sizes, int n_in,
                              void* d_out, int out_size, void* d_ws,
                              size_t ws_size, hipStream_t stream) {
  (void)in_sizes; (void)n_in; (void)out_size; (void)ws_size;
  const int*   input_seqs  = (const int*)d_in[0];
  const int*   target_seqs = (const int*)d_in[1];
  const float* enc_emb = (const float*)d_in[2];
  const float* enc_Wih = (const float*)d_in[3];
  const float* enc_Whh = (const float*)d_in[4];
  const float* enc_bih = (const float*)d_in[5];
  const float* enc_bhh = (const float*)d_in[6];
  const float* dec_emb = (const float*)d_in[7];
  const float* dec_Wih = (const float*)d_in[8];
  const float* dec_Whh = (const float*)d_in[9];
  const float* dec_bih = (const float*)d_in[10];
  const float* dec_bhh = (const float*)d_in[11];
  const float* out_W   = (const float*)d_in[12];
  const float* out_b   = (const float*)d_in[13];
  float* out = (float*)d_out;

  char* ws = (char*)d_ws;
  size_t off = 0;
  auto take = [&](size_t bytes) -> char* {
    char* p = ws + off;
    off += (bytes + 255) & ~(size_t)255;
    return p;
  };
  unsigned* bar    = (unsigned*)take(256);
  __bf16*   encx   = (__bf16*)take((size_t)Tt * Bb * Hh * 2);          // 2 MB
  __bf16*   rnnin  = (__bf16*)take((size_t)Tt * Bb * 2 * Hh * 2);      // 4 MB
  __bf16*   decout = (__bf16*)take((size_t)Bb * Tt * Hh * 2);          // 2 MB
  float*    xp     = (float*)take((size_t)Tt * Bb * 3 * Hh * 4);      // 12.6 MB
  __bf16*   wihE   = (__bf16*)take((size_t)3 * Hh * Hh * 2);          // 1.6 MB
  __bf16*   wihD   = (__bf16*)take((size_t)3 * Hh * 2 * Hh * 2);      // 3.1 MB
  __bf16*   outWb  = (__bf16*)take((size_t)Vv * Hh * 2);              // 32.8 MB
  __bf16*   hbuf   = (__bf16*)take((size_t)2 * Bb * Hh * 2);
  float*    thought= (float*)take((size_t)Bb * Hh * 4);

  (void)hipMemsetAsync(bar, 0, 256, stream);  // reset cross-WG barrier state

  cvt_k<<<2048, 256, 0, stream>>>(enc_Wih, wihE, 3 * Hh * Hh);
  cvt_k<<<2048, 256, 0, stream>>>(dec_Wih, wihD, 3 * Hh * 2 * Hh);
  cvt_k<<<4096, 256, 0, stream>>>(out_W, outWb, Vv * Hh);

  // encoder
  embed_enc_k<<<(Tt * Bb * Hh) / 256, 256, 0, stream>>>(
      input_seqs, enc_emb, encx, Tt * Bb * Hh);
  dim3 gXp(12, 16);  // N=1536/128, M=2048/128
  gemm_bf16_k<<<gXp, 256, 0, stream>>>(encx, wihE, enc_bih, xp,
                                       Tt * Bb, 3 * Hh, Hh);
  gru_scan_k<<<32, 192, 0, stream>>>(xp, enc_Whh, enc_bhh, hbuf, thought,
                                     nullptr, bar, bar + 16, Tt);
  // decoder
  concat_relu_k<<<(Tt * Bb * 2 * Hh) / 256, 256, 0, stream>>>(
      target_seqs, dec_emb, thought, rnnin, Tt * Bb * 2 * Hh);
  gemm_bf16_k<<<gXp, 256, 0, stream>>>(rnnin, wihD, dec_bih, xp,
                                       Tt * Bb, 3 * Hh, 2 * Hh);
  gru_scan_k<<<32, 192, 0, stream>>>(xp, dec_Whh, dec_bhh, hbuf, nullptr,
                                     decout, bar, bar + 16, Tt);
  // vocab projection + log_softmax
  dim3 gLg(Vv / 128, (Bb * Tt) / 128);  // (250, 16)
  gemm_bf16_k<<<gLg, 256, 0, stream>>>(decout, outWb, out_b, out,
                                       Bb * Tt, Vv, Hh);
  logsoftmax_k<<<Bb * Tt, 256, 0, stream>>>(out, Vv);
}
